// Quantize_18373870092610
// MI455X (gfx1250) — matrix operand order repaired and checked by reading, not compiled, not run on hardware
//
#include <hip/hip_runtime.h>
#include <hip/hip_bf16.h>

// ---------------------------------------------------------------------------
// VQ-VAE quantize + EMA update for MI455X (gfx1250, wave32, WMMA).
// Fused [65536x256]@[256x4096] GEMM + per-row argmin using
// v_wmma_f32_16x16x32_bf16 with a 3xBF16 split (~fp32 accuracy).
// v2: double-buffered single-barrier B pipeline + packed b32 LDS stores.
// ---------------------------------------------------------------------------

typedef __attribute__((ext_vector_type(16))) __bf16        v16bf;
typedef __attribute__((ext_vector_type(8)))  float         v8f;
typedef __attribute__((ext_vector_type(4)))  unsigned int  v4u;

#define N_ROWS 65536   // 16*64*64
#define DIM    256
#define NCODE  4096
#define DECAY  0.99f
#define EPSV   1e-5f

// Output offsets (floats), in reference tuple order
#define QUANT_OFF   ((size_t)0)             // 16777216
#define DIFF_OFF    ((size_t)16777216)      // 1
#define IND_OFF     ((size_t)16777217)      // 65536
#define NEWEMB_OFF  ((size_t)16842753)      // 1048576
#define NEWCS_OFF   ((size_t)17891329)      // 4096
#define NEWAVG_OFF  ((size_t)17895425)      // 1048576

// Workspace offsets (floats)
#define WS_NORM  ((size_t)0)        // 4096   ||e_k||^2
#define WS_NT    ((size_t)4096)     // 4096   n_total accumulators
#define WS_ESUM  ((size_t)8192)     // 1048576 embed_sum accumulators [D][K]
#define WS_DSUM  ((size_t)1056768)  // 1      sum of squared diffs
#define WS_NVAL  ((size_t)1056769)  // 1      n = sum(new_cluster_size)
#define WS_ZERO_START 4096
#define WS_ZERO_COUNT (4096 + 1048576 + 2)

#define LDB 264      // padded LDS row stride (bf16 elems): 528B = 16B-aligned,
                     // 132 dwords == 4 mod 64 banks -> conflict-free gathers
#define BTILE (16 * LDB)

__device__ __forceinline__ unsigned short f2bf(float f) {
  unsigned int u = __float_as_uint(f);
  unsigned int r = u + 0x7FFFu + ((u >> 16) & 1u);   // round-to-nearest-even
  return (unsigned short)(r >> 16);
}
__device__ __forceinline__ float b2f(unsigned short h) {
  return __uint_as_float(((unsigned int)h) << 16);
}

// --------------------------- kernel 0: zero ws ------------------------------
__global__ void vq_zero(float* __restrict__ ws) {
  size_t i = (size_t)blockIdx.x * blockDim.x + threadIdx.x;
  if (i < (size_t)WS_ZERO_COUNT) ws[WS_ZERO_START + i] = 0.0f;
}

// ----------------------- kernel 1: codebook norms ---------------------------
__global__ void vq_norms(const float* __restrict__ embed, float* __restrict__ ws) {
  int k = blockIdx.x * 256 + threadIdx.x;   // 16 blocks * 256 = 4096
  float s = 0.0f;
  for (int d = 0; d < DIM; ++d) {
    float v = embed[(size_t)d * NCODE + k];
    s += v * v;
  }
  ws[WS_NORM + k] = s;
}

// -------------- kernel 2: fused GEMM + argmin + gather + stats --------------
__global__ __launch_bounds__(256) void vq_main(const float* __restrict__ flat,
                                               const float* __restrict__ embed,
                                               float* __restrict__ out,
                                               float* __restrict__ ws) {
  // Double-buffered B tile: [2][16 cols][264 dims] bf16, hi and lo splits.
  __shared__ __align__(16) unsigned short sBhi[2][BTILE];
  __shared__ __align__(16) unsigned short sBlo[2][BTILE];
  __shared__ float redV[256 * 8];
  __shared__ int   redI[256 * 8];
  __shared__ int   bestk[128];
  __shared__ float sdiff[256];

  const int t    = threadIdx.x;
  const int lane = t & 31;
  const int wv   = t >> 5;            // wave id 0..7
  const int m    = lane & 15;         // A row within wave tile / B column
  const int ksel = (lane >> 4) << 3;  // A K select: 0 or 8
  const int koff = (lane >> 4) << 4;  // B K select: 0 or 16
  const int blk  = blockIdx.x;
  const size_t myRow = (size_t)blk * 128 + wv * 16 + m;

  // Stage partition: thread owns dims {d0, d0+1} x codes {ksub..ksub+7}
  const int d0   = (t & 127) * 2;
  const int ksub = (t >> 7) * 8;

  // ----- load this wave's 16-row A tile into VGPRs as bf16 hi/lo fragments --
  // A layout (16x32 bf16): lanes 0-15 hold K {0..15}, lanes 16-31 hold
  // K {16..31} of each 32-wide step; 8 steps cover D=256.
  v16bf aH[8], aL[8];
  {
    const float* rp = flat + myRow * DIM;
#pragma unroll
    for (int s = 0; s < 8; ++s) {
      union { unsigned short u[16]; v16bf v; } H, L;
#pragma unroll
      for (int i = 0; i < 8; ++i) {
        float x0 = rp[s * 32 + koff + i];
        float x1 = rp[s * 32 + koff + 8 + i];
        unsigned short h0 = f2bf(x0), h1 = f2bf(x1);
        H.u[i]     = h0;
        H.u[8 + i] = h1;
        L.u[i]     = f2bf(x0 - b2f(h0));
        L.u[8 + i] = f2bf(x1 - b2f(h1));
      }
      aH[s] = H.v;
      aL[s] = L.v;
    }
  }

  float minv[8];
  int   mini[8];
#pragma unroll
  for (int j = 0; j < 8; ++j) { minv[j] = 3.4e38f; mini[j] = 0; }

  const float* dNorm = ws + WS_NORM;

  // ---- prologue: stage k-tile 0 into buffer 0 ------------------------------
  float x0r[8], x1r[8];
  {
    const float* p0 = embed + (size_t)d0 * NCODE + ksub;   // kt = 0
    const float* p1 = p0 + NCODE;
#pragma unroll
    for (int i = 0; i < 8; ++i) { x0r[i] = p0[i]; x1r[i] = p1[i]; }
#pragma unroll
    for (int k = 0; k < 8; ++k) {
      unsigned short h0 = f2bf(x0r[k]), h1 = f2bf(x1r[k]);
      unsigned short l0 = f2bf(x0r[k] - b2f(h0)), l1 = f2bf(x1r[k] - b2f(h1));
      *(unsigned int*)&sBhi[0][(ksub + k) * LDB + d0] =
          (unsigned int)h0 | ((unsigned int)h1 << 16);
      *(unsigned int*)&sBlo[0][(ksub + k) * LDB + d0] =
          (unsigned int)l0 | ((unsigned int)l1 << 16);
    }
  }
  __syncthreads();

  // ---- main pipelined loop over 256 k-tiles --------------------------------
  for (int kt = 0; kt < NCODE / 16; ++kt) {
    const int cur = kt & 1;
    const int nxt = cur ^ 1;

    // prefetch next tile's global data into registers (waits land post-WMMA)
    if (kt + 1 < NCODE / 16) {
      const float* p0 = embed + (size_t)d0 * NCODE + (kt + 1) * 16 + ksub;
      const float* p1 = p0 + NCODE;
#pragma unroll
      for (int i = 0; i < 8; ++i) { x0r[i] = p0[i]; x1r[i] = p1[i]; }
    }

    // compute on current buffer
    v8f c = {0.f, 0.f, 0.f, 0.f, 0.f, 0.f, 0.f, 0.f};
#pragma unroll
    for (int s = 0; s < 8; ++s) {
      const unsigned short* ph = &sBhi[cur][m * LDB + s * 32 + koff];
      const unsigned short* pl = &sBlo[cur][m * LDB + s * 32 + koff];
      union { v4u q[2]; v16bf v; } BH, BL;
      BH.q[0] = *(const v4u*)(ph);
      BH.q[1] = *(const v4u*)(ph + 8);
      BL.q[0] = *(const v4u*)(pl);
      BL.q[1] = *(const v4u*)(pl + 8);
      // 3xBF16 split: Ah*Bh + Ah*Bl + Al*Bh  (~fp32 dot product)
      c = __builtin_amdgcn_wmma_f32_16x16x32_bf16(false, aH[s], false, BH.v,
                                                  (short)0, c, false, false);
      c = __builtin_amdgcn_wmma_f32_16x16x32_bf16(false, aH[s], false, BL.v,
                                                  (short)0, c, false, false);
      c = __builtin_amdgcn_wmma_f32_16x16x32_bf16(false, aL[s], false, BH.v,
                                                  (short)0, c, false, false);
    }

    // dist(k) = ||e_k||^2 - 2*score (||x||^2 constant per row -> irrelevant)
    const int   kcol = kt * 16 + m;
    const float ne   = dNorm[kcol];
#pragma unroll
    for (int j = 0; j < 8; ++j) {
      float dist = ne - 2.0f * c[j];
      if (dist < minv[j]) { minv[j] = dist; mini[j] = kcol; }
    }

    // convert + store next tile into the other buffer (safe: same-iteration
    // readers only touch buf[cur]; iteration boundary barrier below)
    if (kt + 1 < NCODE / 16) {
#pragma unroll
      for (int k = 0; k < 8; ++k) {
        unsigned short h0 = f2bf(x0r[k]), h1 = f2bf(x1r[k]);
        unsigned short l0 = f2bf(x0r[k] - b2f(h0)), l1 = f2bf(x1r[k] - b2f(h1));
        *(unsigned int*)&sBhi[nxt][(ksub + k) * LDB + d0] =
            (unsigned int)h0 | ((unsigned int)h1 << 16);
        *(unsigned int*)&sBlo[nxt][(ksub + k) * LDB + d0] =
            (unsigned int)l0 | ((unsigned int)l1 << 16);
      }
    }
    __syncthreads();
  }

  // ----- cross-lane / cross-wave argmin reduction ---------------------------
  {
    int base = t * 8;
#pragma unroll
    for (int j = 0; j < 8; ++j) { redV[base + j] = minv[j]; redI[base + j] = mini[j]; }
  }
  __syncthreads();
  if (t < 128) {
    // C layout: lanes 0-15 hold rows 0-7 (vgpr j = row), lanes 16-31 rows 8-15.
    int w2 = t >> 4, mm = t & 15;
    int lb = (mm < 8) ? 0 : 16;
    int j  = mm & 7;
    float bv = 3.4e38f;
    int   bi = 0x7fffffff;
#pragma unroll
    for (int l = 0; l < 16; ++l) {
      int idx = ((w2 << 5) + lb + l) * 8 + j;
      float v = redV[idx];
      int   ii = redI[idx];
      if (v < bv || (v == bv && ii < bi)) { bv = v; bi = ii; }
    }
    bestk[t] = bi;
    out[IND_OFF + (size_t)blk * 128 + t] = (float)bi;    // argmin index
    atomicAdd(ws + WS_NT + bi, 1.0f);                    // n_total
  }
  __syncthreads();

  // ----- gather quantize, straight-through output, diff, embed_sum ----------
  float dacc = 0.0f;
  float* esum = ws + WS_ESUM;
  for (int r = 0; r < 128; ++r) {
    int kb = bestk[r];
    size_t row = (size_t)blk * 128 + r;
    float q  = embed[(size_t)t * NCODE + kb];
    float zv = flat[row * DIM + t];
    out[QUANT_OFF + row * DIM + t] = zv + (q - zv);      // straight-through
    float dd = q - zv;
    dacc += dd * dd;
    atomicAdd(esum + (size_t)t * NCODE + kb, zv);        // segment sum
  }
  sdiff[t] = dacc;
  __syncthreads();
  for (int s = 128; s > 0; s >>= 1) {
    if (t < s) sdiff[t] += sdiff[t + s];
    __syncthreads();
  }
  if (t == 0) atomicAdd(ws + WS_DSUM, sdiff[0]);
}

// --------- kernel 3a: new_cluster_size, n = sum(ncs), diff scalar -----------
__global__ __launch_bounds__(256) void vq_finalA(const float* __restrict__ cluster_size,
                                                 float* __restrict__ out,
                                                 float* __restrict__ ws) {
  __shared__ float red[256];
  int t = threadIdx.x;
  float local = 0.0f;
  for (int i = 0; i < 16; ++i) {
    int k = t * 16 + i;
    float v = cluster_size[k] * DECAY + (1.0f - DECAY) * ws[WS_NT + k];
    out[NEWCS_OFF + k] = v;
    local += v;
  }
  red[t] = local;
  __syncthreads();
  for (int s = 128; s > 0; s >>= 1) {
    if (t < s) red[t] += red[t + s];
    __syncthreads();
  }
  if (t == 0) {
    ws[WS_NVAL] = red[0];
    out[DIFF_OFF] = ws[WS_DSUM] * (1.0f / 16777216.0f);  // mean over z elems
  }
}

// ------------- kernel 3b: new_embed_avg and new_embed -----------------------
__global__ __launch_bounds__(256) void vq_finalB(const float* __restrict__ embed_avg,
                                                 float* __restrict__ out,
                                                 const float* __restrict__ ws) {
  size_t i = (size_t)blockIdx.x * 256 + threadIdx.x;   // 4096*256 = 1048576
  int k = (int)(i & (NCODE - 1));
  float nea = embed_avg[i] * DECAY + (1.0f - DECAY) * ws[WS_ESUM + i];
  out[NEWAVG_OFF + i] = nea;
  float n   = ws[WS_NVAL];
  float ncs = out[NEWCS_OFF + k];
  float csv = (ncs + EPSV) / (n + (float)NCODE * EPSV) * n;
  out[NEWEMB_OFF + i] = nea / csv;
}

// ---------------------------------------------------------------------------
extern "C" void kernel_launch(void* const* d_in, const int* in_sizes, int n_in,
                              void* d_out, int out_size, void* d_ws, size_t ws_size,
                              hipStream_t stream) {
  (void)in_sizes; (void)n_in; (void)out_size; (void)ws_size;
  const float* z          = (const float*)d_in[0];   // [16,64,64,256] -> flat [65536,256]
  const float* embed      = (const float*)d_in[1];   // [256,4096]
  const float* cluster_sz = (const float*)d_in[2];   // [4096]
  const float* embed_avg  = (const float*)d_in[3];   // [256,4096]
  float* out = (float*)d_out;
  float* ws  = (float*)d_ws;

  // 1) zero segment-sum accumulators (harness does not re-zero between runs)
  {
    int n = WS_ZERO_COUNT;
    vq_zero<<<(n + 255) / 256, 256, 0, stream>>>(ws);
  }
  // 2) codebook norms ||e_k||^2
  vq_norms<<<NCODE / 256, 256, 0, stream>>>(embed, ws);
  // 3) fused distance-GEMM (bf16 WMMA, 3x split) + argmin + gather + stats
  vq_main<<<N_ROWS / 128, 256, 0, stream>>>(z, embed, out, ws);
  // 4) EMA finalization
  vq_finalA<<<1, 256, 0, stream>>>(cluster_sz, out, ws);
  vq_finalB<<<(DIM * NCODE) / 256, 256, 0, stream>>>(embed_avg, out, ws);
}
